// DGCNNClassification_56538949485055
// MI455X (gfx1250) — compile-verified, matrix operand order checked by reading, not run on hardware
//
#include <hip/hip_runtime.h>
#include <hip/hip_bf16.h>

typedef __attribute__((ext_vector_type(16))) _Float16 v16h;
typedef __attribute__((ext_vector_type(8)))  float    v8f;

#define BATCH 8
#define NPTS  4096
#define KNN   20
#define BN_INV 0.99999499f   /* 1/sqrt(1+1e-5) */
#define CHUNK 256

// ---------------------------------------------------------------------------
// helpers
// ---------------------------------------------------------------------------
__device__ __forceinline__ void cas_fmax(float* addr, float v) {
  unsigned int* ua = (unsigned int*)addr;
  unsigned int old = *ua;
  while (v > __uint_as_float(old)) {
    unsigned int assumed = old;
    old = atomicCAS(ua, assumed, __float_as_uint(v));
    if (old == assumed) break;
  }
}

// ---------------------------------------------------------------------------
// prep: x [B,3,N] -> xt f32 [B,N,4], ph f16 [B,N,32] (zero pad), norms [B,N]
// ---------------------------------------------------------------------------
__global__ void prep_points_kernel(const float* __restrict__ x,
                                   _Float16* __restrict__ ph,
                                   float* __restrict__ xt,
                                   float* __restrict__ norms) {
  int i = blockIdx.x * blockDim.x + threadIdx.x;           // b*N + n
  if (i >= BATCH * NPTS) return;
  int b = i >> 12, n = i & (NPTS - 1);
  float v0 = x[((size_t)b * 3 + 0) * NPTS + n];
  float v1 = x[((size_t)b * 3 + 1) * NPTS + n];
  float v2 = x[((size_t)b * 3 + 2) * NPTS + n];
  _Float16* p = ph + (size_t)i * 32;
#pragma unroll
  for (int c = 0; c < 32; ++c) p[c] = (_Float16)0.0f;
  p[0] = (_Float16)v0; p[1] = (_Float16)v1; p[2] = (_Float16)v2;
  float* t = xt + (size_t)i * 4;
  t[0] = v0; t[1] = v1; t[2] = v2; t[3] = 0.0f;
  norms[i] = v0 * v0 + v1 * v1 + v2 * v2;
}

// f32 features (column slice of cat) -> f16 copy + squared norms
__global__ void cvt_feat_kernel(const float* __restrict__ fin, int rs,
                                _Float16* __restrict__ fh, int C,
                                float* __restrict__ norms) {
  int i = blockIdx.x * blockDim.x + threadIdx.x;
  if (i >= BATCH * NPTS) return;
  const float* r = fin + (size_t)i * rs;
  _Float16* o = fh + (size_t)i * rs;
  float s = 0.0f;
  for (int c = 0; c < C; ++c) { float v = r[c]; s += v * v; o[c] = (_Float16)v; }
  if (norms) norms[i] = s;
}

// W f32 [O, Cin] -> f16 [O, Kpad] zero padded
__global__ void cvt_w_kernel(const float* __restrict__ w, _Float16* __restrict__ wh,
                             int Cin, int Kpad, int O) {
  int i = blockIdx.x * blockDim.x + threadIdx.x;
  if (i >= O * Kpad) return;
  int o = i / Kpad, k = i - o * Kpad;
  wh[i] = (k < Cin) ? (_Float16)w[(size_t)o * Cin + k] : (_Float16)0.0f;
}

// ---------------------------------------------------------------------------
// KNN: per workgroup, 16 query rows; WMMA inner-product tiles vs all N cols.
// A (query) fragments are hoisted into registers and reused for every column
// tile; only B streams from memory. CK = padded feature dim (mult of 32).
// ---------------------------------------------------------------------------
template <int CK>
__global__ __launch_bounds__(256) void knn_kernel(
    const _Float16* __restrict__ feat, int strh,
    const float* __restrict__ norms, int* __restrict__ idx_out) {
  __shared__ float dist[16][CHUNK];
  __shared__ float candv[256][KNN];
  __shared__ int   candi[256][KNN];

  constexpr int KS = CK / 32;
  const int b = blockIdx.y;
  const int row0 = blockIdx.x * 16;
  const int tid  = threadIdx.x;
  const int lane = tid & 31;
  const int wave = tid >> 5;
  const int ml = lane & 15;
  const int kh = lane >> 4;
  const size_t baseRow = (size_t)b * NPTS + row0;

  // hoisted A fragments (16 query rows), reused for all column tiles
  v16h afrag[KS];
#pragma unroll
  for (int ks = 0; ks < KS; ++ks)
    afrag[ks] = *(const v16h*)(feat + (baseRow + ml) * strh + ks * 32 + kh * 16);

  float topv[KNN]; int topi[KNN];
#pragma unroll
  for (int i = 0; i < KNN; ++i) { topv[i] = -3.0e38f; topi[i] = 0; }

  float myNorm[8];
#pragma unroll
  for (int i = 0; i < 8; ++i) myNorm[i] = norms[baseRow + i + kh * 8];

  for (int ch = 0; ch < NPTS / CHUNK; ++ch) {
    // prefetch next chunk's B rows (one row per thread) -> global_prefetch_b8
    if (ch + 1 < NPTS / CHUNK)
      __builtin_prefetch(feat + ((size_t)b * NPTS + (ch + 1) * CHUNK + tid) * strh, 0, 1);

    // --- distance tiles via WMMA (each wave takes column tiles) ---
    for (int t = wave; t < CHUNK / 16; t += 8) {
      const int col0 = ch * CHUNK + t * 16;
      const _Float16* brow = feat + ((size_t)b * NPTS + col0 + ml) * strh + kh * 16;
      v8f acc = {0.f, 0.f, 0.f, 0.f, 0.f, 0.f, 0.f, 0.f};
#pragma unroll
      for (int ks = 0; ks < KS; ++ks) {
        v16h bv = *(const v16h*)(brow + ks * 32);
        acc = __builtin_amdgcn_wmma_f32_16x16x32_f16(false, afrag[ks], false, bv,
                                                     (short)0, acc, false, false);
      }
      const float xc = norms[(size_t)b * NPTS + col0 + ml];
#pragma unroll
      for (int i = 0; i < 8; ++i) {
        float nd = 2.0f * acc[i] - myNorm[i] - xc;
        dist[i + kh * 8][t * 16 + ml] = nd;
      }
    }
    __syncthreads();
    // --- running top-20 (16 threads per row, registers, fully unrolled) ---
    const int r  = tid >> 4;
    const int c0 = tid & 15;
    for (int c = c0; c < CHUNK; c += 16) {
      float v = dist[r][c];
      int gi = ch * CHUNK + c;
      if (v > topv[KNN - 1]) {
        topv[KNN - 1] = v; topi[KNN - 1] = gi;
#pragma unroll
        for (int s = KNN - 1; s > 0; --s) {
          if (topv[s] > topv[s - 1]) {
            float tv = topv[s]; topv[s] = topv[s - 1]; topv[s - 1] = tv;
            int ti = topi[s]; topi[s] = topi[s - 1]; topi[s - 1] = ti;
          }
        }
      }
    }
    __syncthreads();
  }

#pragma unroll
  for (int i = 0; i < KNN; ++i) { candv[tid][i] = topv[i]; candi[tid][i] = topi[i]; }
  __syncthreads();

  if (tid < 16) {
    const int r = tid;
    int* out = idx_out + ((size_t)b * NPTS + row0 + r) * KNN;
    for (int s = 0; s < KNN; ++s) {
      float best = -3.0e38f; int bi = 0, bt = 0, bj = 0;
      for (int t2 = 0; t2 < 16; ++t2) {
        int src = r * 16 + t2;
#pragma unroll
        for (int j = 0; j < KNN; ++j) {
          float v = candv[src][j];
          if (v > best) { best = v; bi = candi[src][j]; bt = src; bj = j; }
        }
      }
      candv[bt][bj] = -3.4e38f;
      out[s] = bi;
    }
  }
}

// ---------------------------------------------------------------------------
// EdgeConv: 16 nodes x 20 edges = 320 rows = 20 WMMA M-tiles across 8 waves.
// W fragments hoisted per wave; edge feature [nbr-ctr ; ctr] built on the fly
// with vectorized (b128) loads when CIN % 16 == 0. max-over-k via LDS CAS-max.
// ---------------------------------------------------------------------------
template <int CIN, int KPAD>
__global__ __launch_bounds__(256) void edgeconv_kernel(
    const float* __restrict__ fin, int rs_in,
    const int* __restrict__ knn_idx,
    const _Float16* __restrict__ Wh,
    const float* __restrict__ gamma, const float* __restrict__ beta,
    float* __restrict__ fout, int rs_out) {
  __shared__ float outmax[16][16];
  __shared__ int nbr[16][KNN];

  constexpr int KS = KPAD / 32;
  const int b = blockIdx.z;
  const int node0 = blockIdx.x * 16;
  const int o0 = blockIdx.y * 16;
  const int tid = threadIdx.x;
  const int lane = tid & 31;
  const int wave = tid >> 5;

  outmax[tid >> 4][tid & 15] = -3.0e38f;
  for (int e = tid; e < 16 * KNN; e += 256) {
    int nd = e / KNN, j = e - nd * KNN;
    nbr[nd][j] = knn_idx[((size_t)b * NPTS + node0 + nd) * KNN + j];
  }
  __syncthreads();

  const int ml = lane & 15;
  const int kh = lane >> 4;

  // hoisted weight (B) fragments: reused across all M-tiles of this wave
  v16h wfrag[KS];
#pragma unroll
  for (int ks = 0; ks < KS; ++ks)
    wfrag[ks] = *(const v16h*)(Wh + (size_t)(o0 + ml) * KPAD + ks * 32 + kh * 16);

  for (int mt = wave; mt < 20; mt += 8) {
    const int row = mt * 16 + ml;              // edge row 0..319
    const int nd = row / KNN;
    const int j  = row - nd * KNN;
    const int nb = nbr[nd][j];
    const float* ctr = fin + ((size_t)b * NPTS + node0 + nd) * rs_in;
    const float* ngh = fin + ((size_t)b * NPTS + nb) * rs_in;

    v8f acc = {0.f, 0.f, 0.f, 0.f, 0.f, 0.f, 0.f, 0.f};
#pragma unroll
    for (int ks = 0; ks < KS; ++ks) {
      const int gkbase = ks * 32 + kh * 16;
      v16h av;
      if constexpr ((CIN % 16) == 0) {
        // 16-wide sub-row lies entirely in one half -> b128 loads
        if (gkbase < CIN) {
#pragma unroll
          for (int q4 = 0; q4 < 4; ++q4) {
            float4 nv = *(const float4*)(ngh + gkbase + q4 * 4);
            float4 cv = *(const float4*)(ctr + gkbase + q4 * 4);
            av[q4 * 4 + 0] = (_Float16)(nv.x - cv.x);
            av[q4 * 4 + 1] = (_Float16)(nv.y - cv.y);
            av[q4 * 4 + 2] = (_Float16)(nv.z - cv.z);
            av[q4 * 4 + 3] = (_Float16)(nv.w - cv.w);
          }
        } else {
          const float* s = ctr + gkbase - CIN;
#pragma unroll
          for (int q4 = 0; q4 < 4; ++q4) {
            float4 cv = *(const float4*)(s + q4 * 4);
            av[q4 * 4 + 0] = (_Float16)cv.x;
            av[q4 * 4 + 1] = (_Float16)cv.y;
            av[q4 * 4 + 2] = (_Float16)cv.z;
            av[q4 * 4 + 3] = (_Float16)cv.w;
          }
        }
      } else {
#pragma unroll
        for (int q = 0; q < 16; ++q) {
          int gk = gkbase + q;
          float v = 0.0f;
          if (gk < CIN)          v = ngh[gk] - ctr[gk];
          else if (gk < 2 * CIN) v = ctr[gk - CIN];
          av[q] = (_Float16)v;
        }
      }
      acc = __builtin_amdgcn_wmma_f32_16x16x32_f16(false, av, false, wfrag[ks],
                                                   (short)0, acc, false, false);
    }
    const int oc = lane & 15;
    const int mh = lane >> 4;
    const float gg = gamma[o0 + oc] * BN_INV;
    const float bb = beta[o0 + oc];
#pragma unroll
    for (int i = 0; i < 8; ++i) {
      int mr = mt * 16 + i + mh * 8;
      int nd2 = mr / KNN;
      float h = acc[i] * gg + bb;
      h = h > 0.0f ? h : 0.2f * h;
      cas_fmax(&outmax[nd2][oc], h);
    }
  }
  __syncthreads();

  {
    int nd = tid >> 4, oc = tid & 15;
    fout[((size_t)b * NPTS + node0 + nd) * rs_out + o0 + oc] = outmax[nd][oc];
  }
}

// ---------------------------------------------------------------------------
// conv5 (512 -> 1024) with fused global max/mean pooling
// pool layout: [B][2048] : amp at 0..1023, aap(sum/N) at 1024..2047
// ---------------------------------------------------------------------------
__global__ void init_pool_kernel(float* __restrict__ pool) {
  int i = blockIdx.x * blockDim.x + threadIdx.x;
  if (i >= BATCH * 2048) return;
  pool[i] = ((i & 2047) < 1024) ? -3.0e38f : 0.0f;
}

__global__ __launch_bounds__(256) void dense_pool_kernel(
    const _Float16* __restrict__ cath, const _Float16* __restrict__ w5h,
    const float* __restrict__ g5, const float* __restrict__ b5,
    float* __restrict__ pool) {
  const int b = blockIdx.z;
  const int n0 = blockIdx.x * 16;
  const int tid = threadIdx.x;
  const int lane = tid & 31;
  const int wave = tid >> 5;
  const int o0 = (blockIdx.y * 8 + wave) * 16;
  const int ml = lane & 15, kh = lane >> 4;

  const _Float16* arow = cath + ((size_t)b * NPTS + n0 + ml) * 512;
  const _Float16* wrow = w5h + (size_t)(o0 + ml) * 512;
  v8f acc = {0.f, 0.f, 0.f, 0.f, 0.f, 0.f, 0.f, 0.f};
#pragma unroll
  for (int k0 = 0; k0 < 512; k0 += 32) {
    v16h av = *(const v16h*)(arow + k0 + kh * 16);
    v16h wv = *(const v16h*)(wrow + k0 + kh * 16);
    acc = __builtin_amdgcn_wmma_f32_16x16x32_f16(false, av, false, wv,
                                                 (short)0, acc, false, false);
  }
  const int oc = lane & 15;
  const float gg = g5[o0 + oc] * BN_INV;
  const float bb = b5[o0 + oc];
  float* amp = pool + (size_t)b * 2048 + o0 + oc;
  float* aap = amp + 1024;
#pragma unroll
  for (int i = 0; i < 8; ++i) {
    float h = acc[i] * gg + bb;
    h = h > 0.0f ? h : 0.2f * h;
    cas_fmax(amp, h);
    atomicAdd(aap, h * (1.0f / (float)NPTS));
  }
}

// ---------------------------------------------------------------------------
// MLP head
// ---------------------------------------------------------------------------
__global__ void lin_bn_lrelu_kernel(const float* __restrict__ in,
                                    const float* __restrict__ w,
                                    const float* __restrict__ bias,
                                    const float* __restrict__ g,
                                    const float* __restrict__ bta,
                                    float* __restrict__ out, int In, int Out) {
  int i = blockIdx.x * blockDim.x + threadIdx.x;
  if (i >= BATCH * Out) return;
  int b = i / Out, o = i - b * Out;
  const float* xr = in + (size_t)b * In;
  const float* wr = w + (size_t)o * In;
  float s = 0.0f;
  for (int k = 0; k < In; ++k) s = fmaf(xr[k], wr[k], s);
  if (bias) s += bias[o];
  s = s * (g[o] * BN_INV) + bta[o];
  out[i] = s > 0.0f ? s : 0.2f * s;
}

__global__ void lin_final_kernel(const float* __restrict__ in,
                                 const float* __restrict__ w,
                                 const float* __restrict__ bias,
                                 float* __restrict__ out, int In, int Out) {
  int i = blockIdx.x * blockDim.x + threadIdx.x;
  if (i >= BATCH * Out) return;
  int b = i / Out, o = i - b * Out;
  const float* xr = in + (size_t)b * In;
  const float* wr = w + (size_t)o * In;
  float s = 0.0f;
  for (int k = 0; k < In; ++k) s = fmaf(xr[k], wr[k], s);
  out[i] = s + bias[o];
}

// ---------------------------------------------------------------------------
// launch
// ---------------------------------------------------------------------------
extern "C" void kernel_launch(void* const* d_in, const int* in_sizes, int n_in,
                              void* d_out, int out_size, void* d_ws, size_t ws_size,
                              hipStream_t stream) {
  (void)in_sizes; (void)n_in; (void)out_size; (void)ws_size;
  const float* x   = (const float*)d_in[0];
  const float* w1  = (const float*)d_in[1];
  const float* g1  = (const float*)d_in[2];
  const float* b1  = (const float*)d_in[3];
  const float* w2  = (const float*)d_in[4];
  const float* g2  = (const float*)d_in[5];
  const float* b2  = (const float*)d_in[6];
  const float* w3  = (const float*)d_in[7];
  const float* g3  = (const float*)d_in[8];
  const float* b3  = (const float*)d_in[9];
  const float* w4  = (const float*)d_in[10];
  const float* g4  = (const float*)d_in[11];
  const float* b4  = (const float*)d_in[12];
  const float* w5  = (const float*)d_in[13];
  const float* g5  = (const float*)d_in[14];
  const float* b5  = (const float*)d_in[15];
  const float* lw1 = (const float*)d_in[16];
  const float* lg1 = (const float*)d_in[17];
  const float* lb1 = (const float*)d_in[18];
  const float* lw2 = (const float*)d_in[19];
  const float* lbs2= (const float*)d_in[20];
  const float* lg2 = (const float*)d_in[21];
  const float* lb2 = (const float*)d_in[22];
  const float* lw3 = (const float*)d_in[23];
  const float* lbs3= (const float*)d_in[24];
  float* out = (float*)d_out;

  // workspace carve (byte offsets)
  char* ws = (char*)d_ws;
  size_t off = 0;
  auto carve = [&](size_t bytes) { void* p = ws + off; off += (bytes + 255) & ~(size_t)255; return p; };
  float*     CAT   = (float*)    carve((size_t)BATCH * NPTS * 512 * 4);
  _Float16*  CATH  = (_Float16*) carve((size_t)BATCH * NPTS * 512 * 2);
  _Float16*  PH    = (_Float16*) carve((size_t)BATCH * NPTS * 32 * 2);
  float*     XT    = (float*)    carve((size_t)BATCH * NPTS * 4 * 4);
  float*     NORMS = (float*)    carve((size_t)BATCH * NPTS * 4);
  int*       IDX   = (int*)      carve((size_t)BATCH * NPTS * KNN * 4);
  _Float16*  W1H   = (_Float16*) carve((size_t)64 * 32 * 2);
  _Float16*  W2H   = (_Float16*) carve((size_t)64 * 128 * 2);
  _Float16*  W3H   = (_Float16*) carve((size_t)128 * 128 * 2);
  _Float16*  W4H   = (_Float16*) carve((size_t)256 * 256 * 2);
  _Float16*  W5H   = (_Float16*) carve((size_t)1024 * 512 * 2);
  float*     POOL  = (float*)    carve((size_t)BATCH * 2048 * 4);
  float*     H1    = (float*)    carve((size_t)BATCH * 512 * 4);
  float*     H2    = (float*)    carve((size_t)BATCH * 256 * 4);

  const int PT = BATCH * NPTS;                      // 32768 points
  // --- prep ---
  prep_points_kernel<<<PT / 256, 256, 0, stream>>>(x, PH, XT, NORMS);
  cvt_w_kernel<<<(64 * 32 + 255) / 256, 256, 0, stream>>>(w1, W1H, 6, 32, 64);
  cvt_w_kernel<<<(64 * 128 + 255) / 256, 256, 0, stream>>>(w2, W2H, 128, 128, 64);
  cvt_w_kernel<<<(128 * 128 + 255) / 256, 256, 0, stream>>>(w3, W3H, 128, 128, 128);
  cvt_w_kernel<<<(256 * 256 + 255) / 256, 256, 0, stream>>>(w4, W4H, 256, 256, 256);
  cvt_w_kernel<<<(1024 * 512 + 255) / 256, 256, 0, stream>>>(w5, W5H, 512, 512, 1024);
  init_pool_kernel<<<(BATCH * 2048 + 255) / 256, 256, 0, stream>>>(POOL);

  dim3 knnGrid(NPTS / 16, BATCH);
  // --- layer 1 (C=3 -> 64) ---
  knn_kernel<32><<<knnGrid, 256, 0, stream>>>(PH, 32, NORMS, IDX);
  edgeconv_kernel<3, 32><<<dim3(NPTS / 16, 64 / 16, BATCH), 256, 0, stream>>>(
      XT, 4, IDX, W1H, g1, b1, CAT + 0, 512);
  // --- layer 2 (64 -> 64) ---
  cvt_feat_kernel<<<PT / 256, 256, 0, stream>>>(CAT + 0, 512, CATH + 0, 64, NORMS);
  knn_kernel<64><<<knnGrid, 256, 0, stream>>>(CATH + 0, 512, NORMS, IDX);
  edgeconv_kernel<64, 128><<<dim3(NPTS / 16, 64 / 16, BATCH), 256, 0, stream>>>(
      CAT + 0, 512, IDX, W2H, g2, b2, CAT + 64, 512);
  // --- layer 3 (64 -> 128) ---
  cvt_feat_kernel<<<PT / 256, 256, 0, stream>>>(CAT + 64, 512, CATH + 64, 64, NORMS);
  knn_kernel<64><<<knnGrid, 256, 0, stream>>>(CATH + 64, 512, NORMS, IDX);
  edgeconv_kernel<64, 128><<<dim3(NPTS / 16, 128 / 16, BATCH), 256, 0, stream>>>(
      CAT + 64, 512, IDX, W3H, g3, b3, CAT + 128, 512);
  // --- layer 4 (128 -> 256) ---
  cvt_feat_kernel<<<PT / 256, 256, 0, stream>>>(CAT + 128, 512, CATH + 128, 128, NORMS);
  knn_kernel<128><<<knnGrid, 256, 0, stream>>>(CATH + 128, 512, NORMS, IDX);
  edgeconv_kernel<128, 256><<<dim3(NPTS / 16, 256 / 16, BATCH), 256, 0, stream>>>(
      CAT + 128, 512, IDX, W4H, g4, b4, CAT + 256, 512);
  // --- complete f16 cat, conv5 + pooling ---
  cvt_feat_kernel<<<PT / 256, 256, 0, stream>>>(CAT + 256, 512, CATH + 256, 256, nullptr);
  dense_pool_kernel<<<dim3(NPTS / 16, 1024 / 128, BATCH), 256, 0, stream>>>(
      CATH, W5H, g5, b5, POOL);
  // --- MLP head ---
  lin_bn_lrelu_kernel<<<(BATCH * 512 + 255) / 256, 256, 0, stream>>>(
      POOL, lw1, nullptr, lg1, lb1, H1, 2048, 512);
  lin_bn_lrelu_kernel<<<(BATCH * 256 + 255) / 256, 256, 0, stream>>>(
      H1, lw2, lbs2, lg2, lb2, H2, 512, 256);
  lin_final_kernel<<<(BATCH * 40 + 255) / 256, 256, 0, stream>>>(
      H2, lw3, lbs3, out, 256, 40);
}
